// DrugGNN_62938450755974
// MI455X (gfx1250) — matrix operand order repaired and checked by reading, not compiled, not run on hardware
//
#include <hip/hip_runtime.h>
#include <hip/hip_bf16.h>

typedef __attribute__((ext_vector_type(16))) _Float16 v16h;
typedef __attribute__((ext_vector_type(8)))  _Float16 v8h;
typedef __attribute__((ext_vector_type(8)))  float    v8f;

#define N_NODES   50000
#define N_EDGES   400000
#define F_IN      78
#define F_IN_PAD  96
#define HID       128
#define G_GRAPHS  1024
#define OUT_DIM   128
#define NEG_SLOPE 0.2f
#define EPS_BN    1e-5f

// ---------- ordered-uint encoding for float atomic max (memset(0) == -inf identity) ----------
__device__ __forceinline__ unsigned f2ord(float f) {
    unsigned u = __float_as_uint(f);
    return (u & 0x80000000u) ? ~u : (u | 0x80000000u);
}
__device__ __forceinline__ float ord2f(unsigned u) {
    return (u & 0x80000000u) ? __uint_as_float(u & 0x7fffffffu) : __uint_as_float(~u);
}

// ---------- conversion kernels ----------
// X[M x K] f32 -> Y[M x Kpad] f16 (zero padded cols)
__global__ void k_convert_pad(const float* __restrict__ X, _Float16* __restrict__ Y,
                              int M, int K, int Kpad) {
    long long t = (long long)blockIdx.x * blockDim.x + threadIdx.x;
    long long total = (long long)M * Kpad;
    if (t >= total) return;
    int k = (int)(t % Kpad);
    long long m = t / Kpad;
    Y[t] = (k < K) ? (_Float16)X[m * K + k] : (_Float16)0.0f;
}

// W[K x Ncol] f32 -> Wt[Ncol x Kpad] f16 (transposed, zero padded)
__global__ void k_convert_wt(const float* __restrict__ W, _Float16* __restrict__ Wt,
                             int K, int Ncol, int Kpad) {
    long long t = (long long)blockIdx.x * blockDim.x + threadIdx.x;
    long long total = (long long)Ncol * Kpad;
    if (t >= total) return;
    int k = (int)(t % Kpad);
    int n = (int)(t / Kpad);
    Wt[t] = (k < K) ? (_Float16)W[(long long)k * Ncol + n] : (_Float16)0.0f;
}

// ---------- WMMA GEMM ----------
// C[M x Ncol] = A[M x K] @ Bt^T.  Each wave computes a 16x64 strip of C:
// one A fragment is reused across 4 B fragments (4 WMMAs per A load -> 4x reuse).
// 4 waves per block cover 4 consecutive M tiles. Ncol must be a multiple of 64,
// K a multiple of 32, M a multiple of 16.
__global__ void __launch_bounds__(128)
k_wmma_gemm(const _Float16* __restrict__ A,   // [M x K] row-major f16
            const _Float16* __restrict__ Bt,  // [Ncol x K] f16 (B transposed)
            float* __restrict__ C,            // [M x Ncol] f32
            _Float16* __restrict__ C16,       // optional f16 copy (same layout)
            const float* __restrict__ bias,   // optional [Ncol]
            int M, int Ncol, int K, int relu) {
    const int wave  = threadIdx.x >> 5;
    const int lane  = threadIdx.x & 31;
    const int tileM = blockIdx.x * 4 + wave;
    if (tileM * 16 >= M) return;            // wave-uniform guard (EXEC all-ones below)
    const int n0    = blockIdx.y * 64;
    const int hb    = lane >> 4;            // 0/1: K-half of fragment
    const int l15   = lane & 15;

    const _Float16* arow = A + (size_t)(tileM * 16 + l15) * K;

    v8f acc[4] = {{}, {}, {}, {}};
#pragma unroll 2
    for (int k0 = 0; k0 < K; k0 += 32) {
        // A fragment (16x32 f16): lane = M row; e<8 -> K=k0+8*hb+e, e>=8 -> K=k0+16+8*hb+(e-8)
        v8h alo = *(const v8h*)(arow + k0 + 8 * hb);
        v8h ahi = *(const v8h*)(arow + k0 + 8 * hb + 16);
        v16h a;
#pragma unroll
        for (int i = 0; i < 8; ++i) { a[i] = alo[i]; a[i + 8] = ahi[i]; }
#pragma unroll
        for (int j = 0; j < 4; ++j) {
            // B fragment (32x16 f16): lane = N col; e<8 -> K=k0+8*hb+e, e>=8 -> K=k0+16+8*hb+(e-8)
            const _Float16* bp = Bt + (size_t)(n0 + 16 * j + l15) * K + k0 + 8 * hb;
            v8h blo = *(const v8h*)bp;
            v8h bhi = *(const v8h*)(bp + 16);
            v16h b;
#pragma unroll
            for (int i = 0; i < 8; ++i) { b[i] = blo[i]; b[i + 8] = bhi[i]; }
            acc[j] = __builtin_amdgcn_wmma_f32_16x16x32_f16(false, a, false, b, (short)0,
                                                            acc[j], false, false);
        }
    }

    // C layout: lane 0-15 col=lane, VGPR r row=r; lanes 16-31 row=r+8
#pragma unroll
    for (int j = 0; j < 4; ++j) {
        const int col = n0 + 16 * j + l15;
        const float bv = bias ? bias[col] : 0.0f;
#pragma unroll
        for (int r = 0; r < 8; ++r) {
            int row = tileM * 16 + r + 8 * hb;
            float v = acc[j][r] + bv;
            if (relu) v = fmaxf(v, 0.0f);
            C[(size_t)row * Ncol + col] = v;
            if (C16) C16[(size_t)row * Ncol + col] = (_Float16)v;
        }
    }
}

// ---------- GAT attention kernels ----------
// per-(node, head) attention scores
__global__ void k_scores(const float* __restrict__ hw, const float* __restrict__ a_s,
                         const float* __restrict__ a_d, float* __restrict__ sc_s,
                         float* __restrict__ sc_d, int n, int H, int C) {
    int t = blockIdx.x * blockDim.x + threadIdx.x;
    if (t >= n * H) return;
    int hd = t % H;
    int nd = t / H;
    const float* row = hw + (size_t)nd * H * C + (size_t)hd * C;
    float ss = 0.0f, sd = 0.0f;
    for (int c = 0; c < C; ++c) {
        float v = row[c];
        ss += v * a_s[hd * C + c];
        sd += v * a_d[hd * C + c];
    }
    sc_s[t] = ss;
    sc_d[t] = sd;
}

__device__ __forceinline__ void edge_pair(const int* ei, int e, int& s, int& d) {
    if (e < N_EDGES) { s = ei[e]; d = ei[N_EDGES + e]; }
    else             { s = d = e - N_EDGES; }   // self loop
}

__device__ __forceinline__ float lrelu(float x) { return x > 0.0f ? x : NEG_SLOPE * x; }

// pass 1: segment max of edge scores into m_ord (ordered-uint)
__global__ void k_edge_max(const int* __restrict__ ei, const float* __restrict__ sc_s,
                           const float* __restrict__ sc_d, unsigned* __restrict__ m_ord,
                           int H) {
    int t = blockIdx.x * blockDim.x + threadIdx.x;
    int total = (N_EDGES + N_NODES) * H;
    if (t >= total) return;
    int hd = t % H, e = t / H, s, d;
    edge_pair(ei, e, s, d);
    float sc = lrelu(sc_s[s * H + hd] + sc_d[d * H + hd]);
    atomicMax(&m_ord[d * H + hd], f2ord(sc));
}

__global__ void k_m_final(const unsigned* __restrict__ m_ord, float* __restrict__ m, int n) {
    int t = blockIdx.x * blockDim.x + threadIdx.x;
    if (t >= n) return;
    float v = ord2f(m_ord[t]);
    m[t] = isfinite(v) ? v : 0.0f;
}

// pass 2: segment sum of exp(e - m)
__global__ void k_edge_denom(const int* __restrict__ ei, const float* __restrict__ sc_s,
                             const float* __restrict__ sc_d, const float* __restrict__ m,
                             float* __restrict__ denom, int H) {
    int t = blockIdx.x * blockDim.x + threadIdx.x;
    int total = (N_EDGES + N_NODES) * H;
    if (t >= total) return;
    int hd = t % H, e = t / H, s, d;
    edge_pair(ei, e, s, d);
    float sc = lrelu(sc_s[s * H + hd] + sc_d[d * H + hd]);
    atomicAdd(&denom[d * H + hd], expf(sc - m[d * H + hd]));
}

// pass 3: weighted scatter-add aggregation, one wave per (edge, head)
__global__ void k_aggregate(const int* __restrict__ ei, const float* __restrict__ sc_s,
                            const float* __restrict__ sc_d, const float* __restrict__ m,
                            const float* __restrict__ denom, const float* __restrict__ hw,
                            float* __restrict__ agg, int H, int C) {
    int pair = blockIdx.x * (blockDim.x >> 5) + (threadIdx.x >> 5);
    int lane = threadIdx.x & 31;
    int total = (N_EDGES + N_NODES) * H;
    if (pair >= total) return;
    int hd = pair % H, e = pair / H, s, d;
    edge_pair(ei, e, s, d);
    float sc = lrelu(sc_s[s * H + hd] + sc_d[d * H + hd]);
    float alpha = expf(sc - m[d * H + hd]) / (denom[d * H + hd] + 1e-16f);
    const float* hrow = hw + (size_t)s * H * C + (size_t)hd * C;
    float* arow = agg + (size_t)d * H * C + (size_t)hd * C;
    for (int c = lane; c < C; c += 32) atomicAdd(&arow[c], alpha * hrow[c]);
}

// bias + batchnorm + relu + optional residual; writes f32 in place + optional f16 copy
__global__ void k_post(float* __restrict__ x, const float* __restrict__ b,
                       const float* __restrict__ g, const float* __restrict__ be,
                       const float* __restrict__ rm, const float* __restrict__ rv,
                       const float* __restrict__ hres, int n, int d, int add_res,
                       _Float16* __restrict__ x16) {
    long long t = (long long)blockIdx.x * blockDim.x + threadIdx.x;
    long long total = (long long)n * d;
    if (t >= total) return;
    int j = (int)(t % d);
    float v = x[t] + b[j];
    v = (v - rm[j]) * (g[j] * rsqrtf(rv[j] + EPS_BN)) + be[j];
    v = fmaxf(v, 0.0f);
    if (add_res) v += hres[t];
    x[t] = v;
    if (x16) x16[t] = (_Float16)v;
}

// ---------- pooling ----------
__global__ void k_pool_accum(const float* __restrict__ h, const int* __restrict__ batch,
                             float* __restrict__ psum, unsigned* __restrict__ pmax,
                             float* __restrict__ counts) {
    long long t = (long long)blockIdx.x * blockDim.x + threadIdx.x;
    long long total = (long long)N_NODES * HID;
    if (t >= total) return;
    int c = (int)(t % HID);
    int nd = (int)(t / HID);
    int b = batch[nd];
    float v = h[t];
    atomicAdd(&psum[b * HID + c], v);
    atomicMax(&pmax[b * HID + c], f2ord(v));
    if (c == 0) atomicAdd(&counts[b], 1.0f);
}

__global__ void k_pool_final(const float* __restrict__ psum, const unsigned* __restrict__ pmax,
                             const float* __restrict__ counts, float* __restrict__ pooled,
                             _Float16* __restrict__ pooled16) {
    int t = blockIdx.x * blockDim.x + threadIdx.x;
    if (t >= G_GRAPHS * HID) return;
    int c = t % HID;
    int gph = t / HID;
    float mean = psum[t] / fmaxf(counts[gph], 1.0f);
    float mx = ord2f(pmax[t]);
    if (!isfinite(mx)) mx = 0.0f;
    pooled[(size_t)gph * 2 * HID + c] = mean;
    pooled[(size_t)gph * 2 * HID + HID + c] = mx;
    pooled16[(size_t)gph * 2 * HID + c] = (_Float16)mean;
    pooled16[(size_t)gph * 2 * HID + HID + c] = (_Float16)mx;
}

// ---------- host side ----------
static inline unsigned cdiv(long long a, long long b) { return (unsigned)((a + b - 1) / b); }

extern "C" void kernel_launch(void* const* d_in, const int* in_sizes, int n_in,
                              void* d_out, int out_size, void* d_ws, size_t ws_size,
                              hipStream_t stream) {
    const float* x      = (const float*)d_in[0];
    const int*   ei     = (const int*)d_in[1];
    const int*   batch  = (const int*)d_in[2];
    const float* W_emb  = (const float*)d_in[3];
    const float* b_emb  = (const float*)d_in[4];
    const float* W[3]  = {(const float*)d_in[5],  (const float*)d_in[13], (const float*)d_in[21]};
    const float* As[3] = {(const float*)d_in[6],  (const float*)d_in[14], (const float*)d_in[22]};
    const float* Ad[3] = {(const float*)d_in[7],  (const float*)d_in[15], (const float*)d_in[23]};
    const float* Bb[3] = {(const float*)d_in[8],  (const float*)d_in[16], (const float*)d_in[24]};
    const float* Gg[3] = {(const float*)d_in[9],  (const float*)d_in[17], (const float*)d_in[25]};
    const float* Be[3] = {(const float*)d_in[10], (const float*)d_in[18], (const float*)d_in[26]};
    const float* Rm[3] = {(const float*)d_in[11], (const float*)d_in[19], (const float*)d_in[27]};
    const float* Rv[3] = {(const float*)d_in[12], (const float*)d_in[20], (const float*)d_in[28]};
    const float* Wf1 = (const float*)d_in[29];
    const float* bf1 = (const float*)d_in[30];
    const float* Wf2 = (const float*)d_in[31];
    const float* bf2 = (const float*)d_in[32];
    float* out = (float*)d_out;

    // workspace layout
    char* base = (char*)d_ws;
    size_t off = 0;
    auto alloc = [&](size_t bytes) -> char* {
        char* p = base + off;
        off = (off + bytes + 255) & ~(size_t)255;
        return p;
    };
    const size_t nb512 = (size_t)N_NODES * 512;
    float*     buf0   = (float*)alloc(nb512 * 4);
    float*     buf1   = (float*)alloc(nb512 * 4);
    float*     buf2   = (float*)alloc(nb512 * 4);
    _Float16*  h16    = (_Float16*)alloc(nb512 * 2);
    _Float16*  x16    = (_Float16*)alloc((size_t)N_NODES * F_IN_PAD * 2);
    _Float16*  wembt  = (_Float16*)alloc((size_t)HID * F_IN_PAD * 2);
    _Float16*  w0t    = (_Float16*)alloc((size_t)512 * 128 * 2);
    _Float16*  w1t    = (_Float16*)alloc((size_t)512 * 512 * 2);
    _Float16*  w2t    = (_Float16*)alloc((size_t)128 * 512 * 2);
    _Float16*  wf1t   = (_Float16*)alloc((size_t)128 * 256 * 2);
    _Float16*  wf2t   = (_Float16*)alloc((size_t)128 * 128 * 2);
    float*     sc_s   = (float*)alloc((size_t)N_NODES * 4 * 4);
    float*     sc_d   = (float*)alloc((size_t)N_NODES * 4 * 4);
    unsigned*  m_ord  = (unsigned*)alloc((size_t)N_NODES * 4 * 4);
    float*     m_buf  = (float*)alloc((size_t)N_NODES * 4 * 4);
    float*     denom  = (float*)alloc((size_t)N_NODES * 4 * 4);
    float*     psum   = (float*)alloc((size_t)G_GRAPHS * HID * 4);
    unsigned*  pmax   = (unsigned*)alloc((size_t)G_GRAPHS * HID * 4);
    float*     counts = (float*)alloc((size_t)G_GRAPHS * 4);
    float*     pooled = (float*)alloc((size_t)G_GRAPHS * 2 * HID * 4);
    _Float16*  pool16 = (_Float16*)alloc((size_t)G_GRAPHS * 2 * HID * 2);
    float*     tmp1   = (float*)alloc((size_t)G_GRAPHS * OUT_DIM * 4);
    _Float16*  tmp116 = (_Float16*)alloc((size_t)G_GRAPHS * OUT_DIM * 2);

    const int TB = 256;

    // --- convert inputs / weights to f16 (transposed, K-padded) ---
    k_convert_pad<<<cdiv((long long)N_NODES * F_IN_PAD, TB), TB, 0, stream>>>(
        x, x16, N_NODES, F_IN, F_IN_PAD);
    k_convert_wt<<<cdiv((long long)HID * F_IN_PAD, TB), TB, 0, stream>>>(W_emb, wembt, F_IN, HID, F_IN_PAD);
    k_convert_wt<<<cdiv(512 * 128, TB), TB, 0, stream>>>(W[0], w0t, 128, 512, 128);
    k_convert_wt<<<cdiv(512 * 512, TB), TB, 0, stream>>>(W[1], w1t, 512, 512, 512);
    k_convert_wt<<<cdiv(128 * 512, TB), TB, 0, stream>>>(W[2], w2t, 512, 128, 512);
    k_convert_wt<<<cdiv(128 * 256, TB), TB, 0, stream>>>(Wf1, wf1t, 256, 128, 256);
    k_convert_wt<<<cdiv(128 * 128, TB), TB, 0, stream>>>(Wf2, wf2t, 128, 128, 128);

    const unsigned mBlocks = cdiv(N_NODES / 16, 4);   // 4 M-tiles (waves) per block
    const unsigned gBlocks = cdiv(G_GRAPHS / 16, 4);

    // --- embedding: h = relu(x @ W_emb + b_emb)  -> buf0 (f32) + h16 ---
    {
        dim3 grid(mBlocks, HID / 64);
        k_wmma_gemm<<<grid, 128, 0, stream>>>(x16, wembt, buf0, h16, b_emb,
                                              N_NODES, HID, F_IN_PAD, 1);
    }

    // --- GAT layers ---
    const int   Hh[3]   = {4, 4, 1};
    const int   din[3]  = {128, 512, 512};
    const int   dout[3] = {512, 512, 128};
    float* hwbuf[3]  = {buf1, buf0, buf0};   // GEMM output  h @ W
    float* aggbuf[3] = {buf2, buf1, buf2};   // aggregation output -> becomes new h
    float* resbuf[3] = {nullptr, buf2, nullptr};
    const int ETOT = N_EDGES + N_NODES;

    for (int i = 0; i < 3; ++i) {
        const int H = Hh[i], C = HID, d = dout[i];
        _Float16* wt = (i == 0) ? w0t : (i == 1) ? w1t : w2t;
        float* hw  = hwbuf[i];
        float* agg = aggbuf[i];

        // hw = h @ W  (h16 holds current h in f16, dims N x din)
        dim3 grid(mBlocks, d / 64);
        k_wmma_gemm<<<grid, 128, 0, stream>>>(h16, wt, hw, nullptr, nullptr,
                                              N_NODES, d, din[i], 0);

        // attention scores per (node, head)
        k_scores<<<cdiv((long long)N_NODES * H, TB), TB, 0, stream>>>(
            hw, As[i], Ad[i], sc_s, sc_d, N_NODES, H, C);

        hipMemsetAsync(m_ord, 0, (size_t)N_NODES * H * 4, stream);   // ord(0) == -inf identity
        hipMemsetAsync(denom, 0, (size_t)N_NODES * H * 4, stream);
        hipMemsetAsync(agg,   0, (size_t)N_NODES * d * 4, stream);

        k_edge_max<<<cdiv((long long)ETOT * H, TB), TB, 0, stream>>>(ei, sc_s, sc_d, m_ord, H);
        k_m_final<<<cdiv((long long)N_NODES * H, TB), TB, 0, stream>>>(m_ord, m_buf, N_NODES * H);
        k_edge_denom<<<cdiv((long long)ETOT * H, TB), TB, 0, stream>>>(ei, sc_s, sc_d, m_buf, denom, H);
        k_aggregate<<<cdiv((long long)ETOT * H, TB / 32), TB, 0, stream>>>(
            ei, sc_s, sc_d, m_buf, denom, hw, agg, H, C);

        // bias + BN + relu (+ residual for layer 1); layer 2 heads==1 so mean == identity
        k_post<<<cdiv((long long)N_NODES * d, TB), TB, 0, stream>>>(
            agg, Bb[i], Gg[i], Be[i], Rm[i], Rv[i], resbuf[i], N_NODES, d,
            (i == 1) ? 1 : 0, (i < 2) ? h16 : nullptr);
    }

    // --- pooling (mean + max over graphs), final h lives in buf2 [N x 128] ---
    hipMemsetAsync(psum,   0, (size_t)G_GRAPHS * HID * 4, stream);
    hipMemsetAsync(pmax,   0, (size_t)G_GRAPHS * HID * 4, stream);
    hipMemsetAsync(counts, 0, (size_t)G_GRAPHS * 4, stream);
    k_pool_accum<<<cdiv((long long)N_NODES * HID, TB), TB, 0, stream>>>(buf2, batch, psum, pmax, counts);
    k_pool_final<<<cdiv(G_GRAPHS * HID, TB), TB, 0, stream>>>(psum, pmax, counts, pooled, pool16);

    // --- MLP head: relu(pooled @ Wf1 + bf1) @ Wf2 + bf2 ---
    {
        dim3 grid1(gBlocks, OUT_DIM / 64);
        k_wmma_gemm<<<grid1, 128, 0, stream>>>(pool16, wf1t, tmp1, tmp116, bf1,
                                               G_GRAPHS, OUT_DIM, 2 * HID, 1);
        dim3 grid2(gBlocks, OUT_DIM / 64);
        k_wmma_gemm<<<grid2, 128, 0, stream>>>(tmp116, wf2t, out, nullptr, bf2,
                                               G_GRAPHS, OUT_DIM, HID, 0);
    }
}
